// MultiDiffHeadAttention_58531814310246
// MI455X (gfx1250) — matrix operand order repaired and checked by reading, not compiled, not run on hardware
//
#include <hip/hip_runtime.h>
#include <cstdint>

typedef __attribute__((ext_vector_type(16))) __bf16 v16bf;
typedef __attribute__((ext_vector_type(8)))  __bf16 v8bf;
typedef __attribute__((ext_vector_type(8)))  float  v8f;
typedef __attribute__((ext_vector_type(4)))  unsigned int u32x4;
typedef __attribute__((ext_vector_type(8)))  int i32x8;
typedef __attribute__((ext_vector_type(4)))  int i32x4;

#define LAMBDA_INIT 0.8f

__device__ __forceinline__ unsigned short f2bf(float f){
  unsigned int x = __float_as_uint(f);
  x = (x + 0x7FFFu + ((x >> 16) & 1u)) >> 16;   // round-to-nearest-even bf16
  return (unsigned short)x;
}

__device__ __forceinline__ v8f wmma_bf16(v16bf a, v16bf b, v8f c){
  // D = A(16x32 bf16) * B(32x16 bf16) + C(16x16 f32)
  return __builtin_amdgcn_wmma_f32_16x16x32_bf16(false, a, false, b, (short)0, c, false, false);
}

// A-fragment: lane holds one row; lanes 0-15 take K 0..15 (c0=0),
// lanes 16-31 take K 16..31 (c0=8). Two 16B loads.
__device__ __forceinline__ v16bf ld_fragA(const unsigned short* row, int c0){
  union { v16bf v; v8bf h[2]; } u;
  u.h[0] = *(const v8bf*)(row + 2*c0);
  u.h[1] = *(const v8bf*)(row + 2*c0 + 8);
  return u.v;
}
// B-fragment: lane holds one column (16 contiguous K values). Two 16B loads.
__device__ __forceinline__ v16bf ld_fragB(const unsigned short* p){
  union { v16bf v; v8bf h[2]; } u;
  u.h[0] = *(const v8bf*)(p);
  u.h[1] = *(const v8bf*)(p + 8);
  return u.v;
}

// ---- Tensor Data Mover: 2-D tile (bf16) from global -> LDS, tracked by TENSORcnt ----
// D# packing per CDNA5 ISA §8: group0 = {count=1, lds_addr, global_addr[56:0], type=2},
// group1 = {data_size=1(2B), tensor_dim0/1, tile_dim0/1, tensor_dim0_stride}; groups 2/3 zero (2-D).
__device__ __forceinline__ void tdm_load_2d(unsigned lds_addr, const void* gptr,
                                            unsigned tensor_d0, unsigned tensor_d1,
                                            unsigned long long stride0,
                                            unsigned tile_d0, unsigned tile_d1){
  unsigned long long ga = (unsigned long long)(uintptr_t)gptr;
  u32x4 g0;
  g0[0] = 1u;                                            // count=1 (valid descriptor)
  g0[1] = lds_addr;                                      // LDS byte address
  g0[2] = (unsigned)(ga & 0xFFFFFFFFull);                // global_addr[31:0]
  g0[3] = (unsigned)((ga >> 32) & 0x1FFFFFFull) | (2u << 30);  // global_addr[56:32] | type=2
  i32x8 g1;
  g1[0] = (int)(1u << 16);                               // data_size=1 -> 2 bytes/elem
  g1[1] = (int)((tensor_d0 & 0xFFFFu) << 16);            // tensor_dim0[15:0]
  g1[2] = (int)(((tensor_d0 >> 16) & 0xFFFFu) | ((tensor_d1 & 0xFFFFu) << 16));
  g1[3] = (int)(((tensor_d1 >> 16) & 0xFFFFu) | ((tile_d0 & 0xFFFFu) << 16));
  g1[4] = (int)(tile_d1 & 0xFFFFu);                      // tile_dim1 | tile_dim2=0
  g1[5] = (int)(stride0 & 0xFFFFFFFFull);                // tensor_dim0_stride[31:0]
  g1[6] = (int)((stride0 >> 32) & 0xFFFFull);            // stride0[47:32] | stride1[15:0]=0
  g1[7] = 0;
  i32x4 gz = {0, 0, 0, 0};
#if __clang_major__ >= 23
  i32x8 gz8 = {0, 0, 0, 0, 0, 0, 0, 0};
  __builtin_amdgcn_tensor_load_to_lds(g0, g1, gz, gz, gz8, 0);
#else
  __builtin_amdgcn_tensor_load_to_lds(g0, g1, gz, gz, 0);
#endif
}

// ---------------- prep: fp32 -> bf16 convert / weight transpose ----------------

__global__ void cvt_bf16_kernel(const float* __restrict__ src,
                                unsigned short* __restrict__ dst, int n){
  for (int i = blockIdx.x*blockDim.x + threadIdx.x; i < n; i += gridDim.x*blockDim.x)
    dst[i] = f2bf(src[i]);
}

// W [Hn, R, C] fp32 -> WT [Hn, C, R] bf16
__global__ void transpose_w_kernel(const float* __restrict__ W,
                                   unsigned short* __restrict__ WT,
                                   int Hn, int R, int C){
  int n = Hn * R * C;
  for (int i = blockIdx.x*blockDim.x + threadIdx.x; i < n; i += gridDim.x*blockDim.x){
    int h   = i / (R*C);
    int rem = i - h*R*C;
    int r   = rem / C;
    int c   = rem - r*C;
    WT[((size_t)h*C + c)*R + r] = f2bf(W[i]);
  }
}

// ---------------- projection GEMM: Out[b,h,t,d] = sum_e X[b,t,e] * WT[h,d,e] ----------------
__global__ __launch_bounds__(256)
void proj_kernel(const unsigned short* __restrict__ X,
                 const unsigned short* __restrict__ WT,
                 unsigned short* __restrict__ Out,
                 int T_, int E_, int D_, int Hn, int transposed){
  const int wave = threadIdx.x >> 5, lane = threadIdx.x & 31;
  const int lo = lane >> 4, li = lane & 15;
  const int row0 = blockIdx.x * 16;
  const int h  = blockIdx.y;
  const int d0 = wave * 16;
  const unsigned short* Arow = X  + (size_t)(row0 + li) * E_;
  const unsigned short* Brow = WT + ((size_t)h * D_ + d0 + li) * E_;
  const int c0 = lo * 8, kbv = lo * 16;
  v8f acc = {};
  for (int e = 0; e < E_; e += 32){
    v16bf a = ld_fragA(Arow + e, c0);
    v16bf b = ld_fragB(Brow + e + kbv);
    acc = wmma_bf16(a, b, acc);
  }
#pragma unroll
  for (int r = 0; r < 8; ++r){
    int rowg = row0 + r + 8*lo;
    int bb = rowg / T_, t = rowg - bb*T_;
    size_t idx = transposed ? (((size_t)(bb*Hn + h)*D_ + (d0 + li))*T_ + t)
                            : (((size_t)(bb*Hn + h)*T_ + t)*(size_t)D_ + d0 + li);
    Out[idx] = f2bf(acc[r]);
  }
}

// ---------------- fused differential flash attention with TDM double buffering ----------------
// grid (T/128, B*H), 256 thr = 8 waves; wave owns a 16-row query tile.
// K-tile (32x128) and V^T-tile (128x32) are DMA'd to LDS by the Tensor Data Mover
// (wave0 -> K, wave1 -> V), double buffered, synced with s_wait_tensorcnt + barrier.
__global__ __launch_bounds__(256)
void diffattn_kernel(const unsigned short* __restrict__ Q,
                     const unsigned short* __restrict__ K,
                     const unsigned short* __restrict__ Vt,
                     const float* __restrict__ lq1, const float* __restrict__ lk1,
                     const float* __restrict__ lq2, const float* __restrict__ lk2,
                     unsigned short* __restrict__ Ctx, int T_){
  __shared__ __align__(16) unsigned short Kst[2][32*128];   // 2 x 8KB
  __shared__ __align__(16) unsigned short Vst[2][128*32];   // 2 x 8KB
  __shared__ __align__(16) unsigned short Pst[8][512];      // per-wave 16x32 bf16 P tile
  const int wave = threadIdx.x >> 5, lane = threadIdx.x & 31;
  const int lo = lane >> 4, li = lane & 15;
  const int bh = blockIdx.y, h = bh & 7, bb = bh >> 3;
  const int q_base = blockIdx.x * 128 + wave * 16;
  const unsigned short* Qb = Q  + (size_t)bh * T_ * 128;
  const unsigned short* Kb = K  + (size_t)bh * T_ * 128;
  const unsigned short* Vb = Vt + (size_t)bh * 128 * T_;
  unsigned short* Pw = &Pst[wave][0];

  // lambda = exp(lq1.lk1) - exp(lq2.lk2) + LAMBDA_INIT (uniform per head; tiny)
  float d1 = 0.f, d2 = 0.f;
  for (int i = 0; i < 64; ++i){
    d1 += lq1[h*64+i]*lk1[h*64+i];
    d2 += lq2[h*64+i]*lk2[h*64+i];
  }
  const float lbd = __expf(d1) - __expf(d2) + LAMBDA_INIT;

  // preload Q A-fragments: hs chunks 0,32 (Q1) and 64,96 (Q2)
  v16bf qf[4];
#pragma unroll
  for (int m = 0; m < 4; ++m){
    const unsigned short* qp = Qb + (size_t)(q_base + li)*128 + m*32;
    qf[m] = ld_fragA(qp, lo*8);
  }

  v8f ctx1[8], ctx2[8];
  float mx1[8], mx2[8], l1[8], l2[8];
#pragma unroll
  for (int f = 0; f < 8; ++f){ ctx1[f] = (v8f){}; ctx2[f] = (v8f){}; }
#pragma unroll
  for (int r = 0; r < 8; ++r){ mx1[r]=mx2[r]=-1e30f; l1[r]=l2[r]=0.f; }

  const float scl = 0.125f;                    // HS^-0.5, HS=64
  const int maxsb = T_/32;
  int nsb_w = (q_base + 16)/32 + 1;            // causal: col <= row+1 (this wave)
  if (nsb_w > maxsb) nsb_w = maxsb;
  int nsb_b = blockIdx.x*4 + 5;                // block-wide bound (wave 7's range)
  if (nsb_b > maxsb) nsb_b = maxsb;

  // prologue: DMA tiles for sb=0 into buffer 0
  if (wave == 0)
    tdm_load_2d((unsigned)(uintptr_t)&Kst[0][0], Kb, 128u, (unsigned)T_, 128ull, 128u, 32u);
  if (wave == 1)
    tdm_load_2d((unsigned)(uintptr_t)&Vst[0][0], Vb, (unsigned)T_, 128u, (unsigned long long)T_, 32u, 128u);
  if (wave < 2) __builtin_amdgcn_s_wait_tensorcnt(0);
  __syncthreads();

  for (int sb = 0; sb < nsb_b; ++sb){
    const int cur = sb & 1;
    const int nxt = (sb + 1) & 1;
    // issue next tile DMA (overlaps with compute below); buffer 'nxt' was last
    // read in iteration sb-1 and released by that iteration's barrier.
    if (sb + 1 < nsb_b){
      const int s1n = (sb + 1) * 32;
      if (wave == 0)
        tdm_load_2d((unsigned)(uintptr_t)&Kst[nxt][0], Kb + (size_t)s1n*128,
                    128u, (unsigned)T_, 128ull, 128u, 32u);
      if (wave == 1)
        tdm_load_2d((unsigned)(uintptr_t)&Vst[nxt][0], Vb + s1n,
                    (unsigned)T_, 128u, (unsigned long long)T_, 32u, 128u);
    }

    if (sb < nsb_w){
      const int s0 = sb * 32;
      // V^T B-fragments from LDS (reused by both streams)
      v16bf vbf[8];
#pragma unroll
      for (int f = 0; f < 8; ++f)
        vbf[f] = ld_fragB(&Vst[cur][(f*16 + li)*32 + lo*16]);
#pragma unroll
      for (int mat = 0; mat < 2; ++mat){
        // scores: 16 q-rows x 32 keys, 4 QK^T WMMAs with K-fragments from LDS
        v8f c0 = {}, c1 = {};
#pragma unroll
        for (int kk = 0; kk < 2; ++kk){
          c0 = wmma_bf16(qf[mat*2+kk], ld_fragB(&Kst[cur][li*128 + mat*64 + kk*32 + lo*16]), c0);
          c1 = wmma_bf16(qf[mat*2+kk], ld_fragB(&Kst[cur][(16+li)*128 + mat*64 + kk*32 + lo*16]), c1);
        }
        float* mx = mat ? mx2 : mx1;
        float* ls = mat ? l2  : l1;
        v8f*  ctx = mat ? ctx2 : ctx1;
#pragma unroll
        for (int r = 0; r < 8; ++r){
          float v0 = c0[r]*scl, v1 = c1[r]*scl;
          const int rowg = q_base + r + 8*lo;
          if (s0 + li      > rowg + 1) v0 = -1e30f;   // causal mask (diag=1)
          if (s0 + 16 + li > rowg + 1) v1 = -1e30f;
          float bm = fmaxf(v0, v1);
#pragma unroll
          for (int msk = 1; msk < 16; msk <<= 1) bm = fmaxf(bm, __shfl_xor(bm, msk, 32));
          const float mn = fmaxf(mx[r], bm);
          const float sc = __expf(mx[r] - mn);
          const float p0 = __expf(v0 - mn), p1 = __expf(v1 - mn);
          float rs = p0 + p1;
#pragma unroll
          for (int msk = 1; msk < 16; msk <<= 1) rs += __shfl_xor(rs, msk, 32);
          ls[r] = ls[r]*sc + rs;
          mx[r] = mn;
#pragma unroll
          for (int f = 0; f < 8; ++f) ctx[f][r] *= sc;
          Pw[(r + 8*lo)*32 + li]      = f2bf(p0);     // C-fragment -> row-major LDS
          Pw[(r + 8*lo)*32 + 16 + li] = f2bf(p1);
        }
        // reload P as an A-fragment (same-wave LDS ops are in-order)
        union { v16bf v; v8bf hh[2]; } ua;
        ua.hh[0] = *(const v8bf*)&Pw[li*32 + lo*16];
        ua.hh[1] = *(const v8bf*)&Pw[li*32 + lo*16 + 8];
#pragma unroll
        for (int f = 0; f < 8; ++f) ctx[f] = wmma_bf16(ua.v, vbf[f], ctx[f]);
      }
    }

    if (sb + 1 < nsb_b && wave < 2) __builtin_amdgcn_s_wait_tensorcnt(0);
    __syncthreads();   // next buffer ready; current buffer released for reuse
  }

  const float og = 1.0f - LAMBDA_INIT;
#pragma unroll
  for (int f = 0; f < 8; ++f){
#pragma unroll
    for (int r = 0; r < 8; ++r){
      const int t = q_base + r + 8*lo;
      float val = (ctx1[f][r]/l1[r] - lbd*(ctx2[f][r]/l2[r])) * og;
      Ctx[((size_t)(bb*T_ + t))*1024 + h*128 + f*16 + li] = f2bf(val);
    }
  }
}

// ---------------- output projection: Out = Ctx @ Wp + bp (fp32 out) ----------------
__global__ __launch_bounds__(256)
void outproj_kernel(const unsigned short* __restrict__ Ctx,
                    const unsigned short* __restrict__ WpT,
                    const float* __restrict__ bp, float* __restrict__ Out,
                    int E_, int Dt){
  const int wave = threadIdx.x >> 5, lane = threadIdx.x & 31;
  const int lo = lane >> 4, li = lane & 15;
  const int row0 = blockIdx.x * 16;
  const int e0 = blockIdx.y * 128 + wave * 16;
  const unsigned short* Arow = Ctx + (size_t)(row0 + li)*Dt;
  const unsigned short* Brow = WpT + (size_t)(e0 + li)*Dt;
  const int c0 = lo*8, kbv = lo*16;
  v8f acc = {};
  for (int k = 0; k < Dt; k += 32)
    acc = wmma_bf16(ld_fragA(Arow + k, c0), ld_fragB(Brow + k + kbv), acc);
  const float bias = bp[e0 + li];
#pragma unroll
  for (int r = 0; r < 8; ++r)
    Out[(size_t)(row0 + r + 8*lo)*E_ + e0 + li] = acc[r] + bias;
}

// ---------------- launch ----------------
extern "C" void kernel_launch(void* const* d_in, const int* in_sizes, int n_in,
                              void* d_out, int out_size, void* d_ws, size_t ws_size,
                              hipStream_t stream) {
  (void)in_sizes; (void)n_in; (void)out_size; (void)ws_size;
  constexpr int B = 2, T = 2048, E = 1024, H = 8, D = 128, Dt = 1024;
  const float* q   = (const float*)d_in[0];
  const float* k   = (const float*)d_in[1];
  const float* v   = (const float*)d_in[2];
  const float* Wq  = (const float*)d_in[3];
  const float* Wk  = (const float*)d_in[4];
  const float* Wv  = (const float*)d_in[5];
  const float* lq1 = (const float*)d_in[6];
  const float* lk1 = (const float*)d_in[7];
  const float* lq2 = (const float*)d_in[8];
  const float* lk2 = (const float*)d_in[9];
  const float* Wp  = (const float*)d_in[10];
  const float* bp  = (const float*)d_in[11];
  float* out = (float*)d_out;

  char* ws = (char*)d_ws;
  size_t off = 0;
  auto alloc = [&](size_t elems)->unsigned short*{
    unsigned short* p = (unsigned short*)(ws + off);
    off += ((elems * sizeof(unsigned short)) + 255) & ~(size_t)255;
    return p;
  };
  unsigned short* qb  = alloc((size_t)B*T*E);
  unsigned short* kb  = alloc((size_t)B*T*E);
  unsigned short* vb  = alloc((size_t)B*T*E);
  unsigned short* WqT = alloc((size_t)H*D*E);
  unsigned short* WkT = alloc((size_t)H*D*E);
  unsigned short* WvT = alloc((size_t)H*D*E);
  unsigned short* WpT = alloc((size_t)Dt*Dt);
  unsigned short* Qp  = alloc((size_t)B*H*T*D);
  unsigned short* Kp  = alloc((size_t)B*H*T*D);
  unsigned short* Vt  = alloc((size_t)B*H*T*D);
  unsigned short* Cx  = alloc((size_t)B*T*Dt);

  const int nact = B*T*E;
  cvt_bf16_kernel<<<1024, 256, 0, stream>>>(q, qb, nact);
  cvt_bf16_kernel<<<1024, 256, 0, stream>>>(k, kb, nact);
  cvt_bf16_kernel<<<1024, 256, 0, stream>>>(v, vb, nact);
  transpose_w_kernel<<<1024, 256, 0, stream>>>(Wq, WqT, H, E, D);
  transpose_w_kernel<<<1024, 256, 0, stream>>>(Wk, WkT, H, E, D);
  transpose_w_kernel<<<1024, 256, 0, stream>>>(Wv, WvT, H, E, D);
  transpose_w_kernel<<<1024, 256, 0, stream>>>(Wp, WpT, 1, Dt, Dt);

  dim3 pg(B*T/16, H);
  proj_kernel<<<pg, 256, 0, stream>>>(qb, WqT, Qp, T, E, D, H, 0);
  proj_kernel<<<pg, 256, 0, stream>>>(kb, WkT, Kp, T, E, D, H, 0);
  proj_kernel<<<pg, 256, 0, stream>>>(vb, WvT, Vt, T, E, D, H, 1);  // V^T [B,H,D,T]

  dim3 ag(T/128, B*H);
  diffattn_kernel<<<ag, 256, 0, stream>>>(Qp, Kp, Vt, lq1, lk1, lq2, lk2, Cx, T);

  dim3 ogr(B*T/16, E/128);
  outproj_kernel<<<ogr, 256, 0, stream>>>(Cx, WpT, bp, out, E, Dt);
}
